// DecoderRNN_65292092834014
// MI455X (gfx1250) — matrix operand order repaired and checked by reading, not compiled, not run on hardware
//
#include <hip/hip_runtime.h>

// ---------------------------------------------------------------- constants
constexpr int NB  = 40;      // batch
constexpr int NL  = 1000;    // encoder length
constexpr int NV  = 78864;   // vocab
constexpr int NE  = 300;     // embed dim
constexpr int NH  = 512;     // hidden
constexpr int NA  = 300;     // attention dim
constexpr int NT  = 20;      // steps
constexpr int START = 2;
constexpr int APAD = 320;    // NA padded to 32 (proj N and vocab K)
constexpr int KVP  = 320;    // vocab GEMM K (== APAD)
constexpr int NBL  = NB * NL;           // 40000
constexpr int PMT  = NBL / 16;          // 2500 M tiles (projection)
constexpr int PNT  = APAD / 16;         // 20 N tiles
constexpr int NBLK = 5;                 // N tiles per wave (projection)
constexpr int NGRP = PNT / NBLK;        // 4 N groups
constexpr int VNT  = NV / 16;           // 4929 N tiles (vocab), exact

// ---------------------------------------------------------------- WMMA types
typedef __attribute__((ext_vector_type(16))) __bf16 v16bf;
typedef __attribute__((ext_vector_type(8)))  __bf16 v8bf;
typedef __attribute__((ext_vector_type(8)))  float  v8f;

__device__ __forceinline__ v16bf ld_frag(const __bf16* p0, const __bf16* p1) {
  v8bf lo = *(const v8bf*)p0;
  v8bf hi = *(const v8bf*)p1;
  return __builtin_shufflevector(lo, hi, 0,1,2,3,4,5,6,7,8,9,10,11,12,13,14,15);
}

__device__ __forceinline__ float sigf(float x) { return 1.0f / (1.0f + __expf(-x)); }

// fp32 dot of a global row against an LDS vector, float4-vectorized (n % 4 == 0)
__device__ __forceinline__ float dot4(const float* __restrict__ w, const float* s, int n) {
  const float4* w4 = (const float4*)w;
  const float4* s4 = (const float4*)s;
  float acc = 0.f;
  int n4 = n >> 2;
  #pragma unroll 4
  for (int i = 0; i < n4; ++i) {
    float4 a = w4[i], b = s4[i];
    acc += a.x*b.x + a.y*b.y + a.z*b.z + a.w*b.w;
  }
  return acc;
}

// block-wide reductions, blockDim.x == 1024
__device__ __forceinline__ float blk_max(float v, float* s_red, int tid) {
  s_red[tid] = v; __syncthreads();
  for (int st = 512; st > 0; st >>= 1) { if (tid < st) s_red[tid] = fmaxf(s_red[tid], s_red[tid+st]); __syncthreads(); }
  float r = s_red[0]; __syncthreads();
  return r;
}
__device__ __forceinline__ float blk_sum(float v, float* s_red, int tid) {
  s_red[tid] = v; __syncthreads();
  for (int st = 512; st > 0; st >>= 1) { if (tid < st) s_red[tid] += s_red[tid+st]; __syncthreads(); }
  float r = s_red[0]; __syncthreads();
  return r;
}

// ---------------------------------------------------------------- conversions
__global__ void cvt_bf16_kernel(const float* __restrict__ src, __bf16* __restrict__ dst, size_t n) {
  size_t i = (size_t)blockIdx.x * blockDim.x + threadIdx.x;
  size_t stride = (size_t)gridDim.x * blockDim.x;
  for (; i < n; i += stride) dst[i] = (__bf16)src[i];
}

// pack 5 [300,512] weights into bf16 [5][320][512] (rows 300..319 zero)
__global__ void pack_w5_kernel(const float* __restrict__ w0, const float* __restrict__ w1,
                               const float* __restrict__ w2, const float* __restrict__ w3,
                               const float* __restrict__ w4, __bf16* __restrict__ dst) {
  size_t n = (size_t)5 * APAD * NH;
  size_t i = (size_t)blockIdx.x * blockDim.x + threadIdx.x;
  size_t stride = (size_t)gridDim.x * blockDim.x;
  for (; i < n; i += stride) {
    int k = (int)(i % NH);
    int r = (int)((i / NH) % APAD);
    int w = (int)(i / ((size_t)NH * APAD));
    const float* W = (w == 0) ? w0 : (w == 1) ? w1 : (w == 2) ? w2 : (w == 3) ? w3 : w4;
    dst[i] = (r < NA) ? (__bf16)W[(size_t)r * NH + k] : (__bf16)0.f;
  }
}

// out_W [78864,300] -> bf16 [78864,320] (K padded with zeros)
__global__ void cvt_outw_kernel(const float* __restrict__ src, __bf16* __restrict__ dst) {
  size_t n = (size_t)NV * KVP;
  size_t i = (size_t)blockIdx.x * blockDim.x + threadIdx.x;
  size_t stride = (size_t)gridDim.x * blockDim.x;
  for (; i < n; i += stride) {
    int k = (int)(i % KVP);
    int v = (int)(i / KVP);
    dst[i] = (k < NA) ? (__bf16)src[(size_t)v * NA + k] : (__bf16)0.f;
  }
}

// ---------------------------------------------------------------- projection GEMM (WMMA)
// projb[w][bl][a] = encb[bl,:] . W_w[a,:] (+bias) ; w==4 -> tanh. bf16 out.
// Each wave: one (w, m-tile) x 5 consecutive N tiles -> A fragment reused 5x.
__global__ __launch_bounds__(256)
void proj_gemm_kernel(const __bf16* __restrict__ encb, const __bf16* __restrict__ w5b,
                      const float* __restrict__ b0, const float* __restrict__ b1,
                      const float* __restrict__ b2, const float* __restrict__ b3,
                      const float* __restrict__ b4, __bf16* __restrict__ projb) {
  const int lane = threadIdx.x & 31;
  const int wid  = threadIdx.x >> 5;
  int id = blockIdx.x * 8 + wid;
  if (id >= 5 * PMT * NGRP) return;
  int w   = id / (PMT * NGRP);
  int rem = id % (PMT * NGRP);
  int mt  = rem / NGRP;
  int ng  = rem % NGRP;
  int nt0 = ng * NBLK;
  const int lh = lane & 15;
  const int hb = lane >> 4;   // lane-half select
  v8f acc[NBLK];
  #pragma unroll
  for (int j = 0; j < NBLK; ++j) acc[j] = (v8f){0.f,0.f,0.f,0.f,0.f,0.f,0.f,0.f};
  const __bf16* arow = encb + ((size_t)(mt*16 + lh)) * NH + hb*8;
  const __bf16* brow[NBLK];
  #pragma unroll
  for (int j = 0; j < NBLK; ++j)
    brow[j] = w5b + ((size_t)(w*APAD + (nt0 + j)*16 + lh)) * NH + hb*8;
  #pragma unroll 2
  for (int k0 = 0; k0 < NH; k0 += 32) {
    __builtin_prefetch(arow + k0 + 128, 0, 1);
    v16bf fa = ld_frag(arow + k0, arow + k0 + 16);
    #pragma unroll
    for (int j = 0; j < NBLK; ++j) {
      v16bf fb = ld_frag(brow[j] + k0, brow[j] + k0 + 16);
      acc[j] = __builtin_amdgcn_wmma_f32_16x16x32_bf16(false, fa, false, fb, (short)0, acc[j], false, false);
    }
  }
  const float* bias = (w == 0) ? b0 : (w == 1) ? b1 : (w == 2) ? b2 : (w == 3) ? b3 : b4;
  #pragma unroll
  for (int j = 0; j < NBLK; ++j) {
    int n = (nt0 + j)*16 + lh;
    float bv = (n < NA) ? bias[n] : 0.f;
    #pragma unroll
    for (int r = 0; r < 8; ++r) {
      int m = mt*16 + r + hb*8;
      float v = acc[j][r] + bv;
      if (w == 4) v = tanhf(v);
      if (n >= NA) v = 0.f;
      projb[((size_t)w * NBL + m) * APAD + n] = (__bf16)v;
    }
  }
}

// ---------------------------------------------------------------- vocab GEMM (WMMA)
// dout[b,t,v] = attA[b,:] . out_W[v,:] + out_b[v]
// Each wave: one N tile x all 3 M tiles -> B fragment (the big operand) reused 3x.
__global__ __launch_bounds__(256)
void vocab_gemm_kernel(const __bf16* __restrict__ attA, const __bf16* __restrict__ outWb,
                       const float* __restrict__ out_b, float* __restrict__ dout, int t) {
  const int lane = threadIdx.x & 31;
  const int wid  = threadIdx.x >> 5;
  int nt = blockIdx.x * 8 + wid;
  if (nt >= VNT) return;
  const int lh = lane & 15;
  const int hb = lane >> 4;
  v8f acc[3];
  #pragma unroll
  for (int m = 0; m < 3; ++m) acc[m] = (v8f){0.f,0.f,0.f,0.f,0.f,0.f,0.f,0.f};
  const __bf16* brow = outWb + ((size_t)(nt*16 + lh)) * KVP + hb*8;
  const __bf16* arow[3];
  #pragma unroll
  for (int m = 0; m < 3; ++m) arow[m] = attA + ((size_t)(m*16 + lh)) * KVP + hb*8;
  #pragma unroll
  for (int k0 = 0; k0 < KVP; k0 += 32) {
    __builtin_prefetch(brow + k0 + 128, 0, 1);
    v16bf fb = ld_frag(brow + k0, brow + k0 + 16);
    #pragma unroll
    for (int m = 0; m < 3; ++m) {
      v16bf fa = ld_frag(arow[m] + k0, arow[m] + k0 + 16);
      acc[m] = __builtin_amdgcn_wmma_f32_16x16x32_bf16(false, fa, false, fb, (short)0, acc[m], false, false);
    }
  }
  int n = nt*16 + lh;
  float bv = out_b[n];
  #pragma unroll
  for (int mtile = 0; mtile < 3; ++mtile) {
    #pragma unroll
    for (int r = 0; r < 8; ++r) {
      int m = mtile*16 + r + hb*8;
      if (m < NB) dout[((size_t)m * NT + t) * NV + n] = acc[mtile][r] + bv;
    }
  }
}

// ---------------------------------------------------------------- embedding gather
__global__ void embed_kernel(const int* __restrict__ inputs, const float* __restrict__ embed,
                             float* __restrict__ emb, int t) {
  int b = blockIdx.x, tid = threadIdx.x;
  int sym = (t == 0) ? START : inputs[b * NT + (t - 1)];
  for (int e = tid; e < NE; e += 256) emb[b * APAD + e] = embed[(size_t)sym * NE + e];
}

// ---------------------------------------------------------------- generic LSTM cell
// one workgroup per batch row; 1024 threads; 2 gates per thread
__global__ __launch_bounds__(1024)
void lstm_kernel(const float* __restrict__ xsrc, int xstride, int xdim,
                 const float* __restrict__ Wih, const float* __restrict__ Whh,
                 const float* __restrict__ bias,
                 float* __restrict__ hbuf, float* __restrict__ cbuf) {
  int b = blockIdx.x, tid = threadIdx.x;
  __shared__ __align__(16) float s_x[NH];
  __shared__ __align__(16) float s_h[NH];
  __shared__ float s_g[4 * NH];
  if (tid < xdim) s_x[tid] = xsrc[(size_t)b * xstride + tid];
  if (tid >= 512) s_h[tid - 512] = hbuf[(size_t)b * NH + (tid - 512)];
  __syncthreads();
  int j  = tid & 511;
  int g0 = (tid >> 9) << 1;
  #pragma unroll
  for (int gi = 0; gi < 2; ++gi) {
    int g = g0 + gi;
    int row = g * NH + j;
    float acc = bias[row];
    acc += dot4(Wih + (size_t)row * xdim, s_x, xdim);
    acc += dot4(Whh + (size_t)row * NH,  s_h, NH);
    s_g[g * NH + j] = acc;
  }
  __syncthreads();
  if (tid < NH) {
    float gi = sigf(s_g[tid]);
    float gf = sigf(s_g[NH + tid]);
    float gg = tanhf(s_g[2 * NH + tid]);
    float go = sigf(s_g[3 * NH + tid]);
    float c = gf * cbuf[(size_t)b * NH + tid] + gi * gg;
    float h = go * tanhf(c);
    cbuf[(size_t)b * NH + tid] = c;
    hbuf[(size_t)b * NH + tid] = h;
  }
}

// ---------------------------------------------------------------- dec_in projection
__global__ __launch_bounds__(512)
void decin_kernel(const float* __restrict__ emb, const float* __restrict__ eh2,
                  const float* __restrict__ attc, const float* __restrict__ selb,
                  const float* __restrict__ in_W, const float* __restrict__ in_b,
                  float* __restrict__ decin, int t) {
  int b = blockIdx.x, tid = threadIdx.x;
  __shared__ __align__(16) float s_cat[1624];
  for (int i = tid; i < 1624; i += 512) {
    float v;
    if      (i < 300)  v = emb [b * APAD + i];
    else if (i < 812)  v = eh2 [(size_t)b * NH + (i - 300)];
    else if (i < 1112) v = attc[b * APAD + (i - 812)];
    else               v = selb[(size_t)b * NH + (i - 1112)];
    s_cat[i] = v;
  }
  __syncthreads();
  int j = tid;
  float acc = in_b[j] + dot4(in_W + (size_t)j * 1624, s_cat, 1624);
  int i2 = j & ~1;
  float ang = (float)t * __powf(10000.0f, -(float)i2 / 512.0f);
  acc += (j & 1) ? cosf(ang) : sinf(ang);
  decin[(size_t)b * NH + j] = acc;
}

// ---------------------------------------------------------------- attention
__global__ __launch_bounds__(1024)
void attn_kernel(const float* __restrict__ hsrc, const float* __restrict__ Wq,
                 const float* __restrict__ bq,
                 const __bf16* __restrict__ projK, const __bf16* __restrict__ projV,
                 const unsigned char* __restrict__ mask,
                 const float* __restrict__ addsrc, float* __restrict__ outb) {
  int b = blockIdx.x, tid = threadIdx.x;
  __shared__ __align__(16) float s_h[NH];
  __shared__ float s_q[APAD];
  __shared__ float s_p[1024];
  __shared__ float s_red[1024];
  if (tid < NH) s_h[tid] = hsrc[(size_t)b * NH + tid];
  __syncthreads();
  if (tid < NA) s_q[tid] = bq[tid] + dot4(Wq + (size_t)tid * NH, s_h, NH);
  __syncthreads();
  float sc = -1e30f;
  if (tid < NL) {
    const __bf16* kp = projK + (size_t)(b * NL + tid) * APAD;
    float s = 0.f;
    #pragma unroll 4
    for (int a = 0; a < NA; ++a) s += s_q[a] * (float)kp[a];
    sc = mask[b * NL + tid] ? -1e9f : s;
  }
  float mx = blk_max(sc, s_red, tid);
  float e  = (tid < NL) ? __expf(sc - mx) : 0.f;
  float sm = blk_sum(e, s_red, tid);
  s_p[tid] = e / sm;
  __syncthreads();
  if (tid < NA) {
    const __bf16* vp = projV + (size_t)(b * NL) * APAD + tid;
    float acc = 0.f;
    for (int l = 0; l < NL; ++l) acc += s_p[l] * (float)vp[(size_t)l * APAD];
    if (addsrc) acc += addsrc[b * APAD + tid];
    outb[b * APAD + tid] = acc;
  }
}

// ---------------------------------------------------------------- copy scores, sel, attA
__global__ __launch_bounds__(1024)
void copy_sel_kernel(const float* __restrict__ attc, const __bf16* __restrict__ copyK,
                     const unsigned char* __restrict__ mask, const int* __restrict__ enc_in,
                     const int* __restrict__ inputs, const __bf16* __restrict__ encb,
                     const float* __restrict__ scale_p, float* __restrict__ score_e,
                     float* __restrict__ selb, __bf16* __restrict__ attA, int t) {
  int b = blockIdx.x, tid = threadIdx.x;
  __shared__ float s_att[APAD];
  __shared__ float s_p[1024];
  __shared__ float s_red[1024];
  if (tid < APAD) s_att[tid] = (tid < NA) ? attc[b * APAD + tid] : 0.f;
  __syncthreads();
  float sc = -1e30f;
  if (tid < NL) {
    const __bf16* kp = copyK + (size_t)(b * NL + tid) * APAD;
    float s = 0.f;
    #pragma unroll 4
    for (int a = 0; a < NA; ++a) s += s_att[a] * (float)kp[a];
    sc = mask[b * NL + tid] ? -1e9f : s;
  }
  float mx = blk_max(sc, s_red, tid);
  float e  = (tid < NL) ? __expf(sc - mx) : 0.f;
  float sm = blk_sum(e, s_red, tid);
  float p  = e / sm;                              // score_c
  float scl = scale_p[0];
  int   so  = inputs[b * NT + t];
  float sf  = 0.f;
  if (tid < NL) {
    score_e[b * 1024 + tid] = p * scl * scl;
    sf = (enc_in[b * NL + tid] == so) ? p : 0.f;  // score_f (unnormalized)
  }
  s_p[tid] = sf;
  float fs = blk_sum(sf, s_red, tid);
  float inv = 1.0f / (fs + 1e-8f);
  __syncthreads();
  if (tid < NH) {
    const __bf16* ep = encb + (size_t)(b * NL) * NH + tid;
    float acc = 0.f;
    for (int l = 0; l < NL; ++l) acc += s_p[l] * (float)ep[(size_t)l * NH];
    selb[(size_t)b * NH + tid] = acc * inv;
  }
  if (tid < KVP) attA[b * KVP + tid] = (tid < NA) ? (__bf16)s_att[tid] : (__bf16)0.f;
}

// ---------------------------------------------------------------- scatter-add of copy scores
__global__ void scatter_kernel(const int* __restrict__ enc_in, const float* __restrict__ score_e,
                               float* __restrict__ dout, int t) {
  int b = blockIdx.x, tid = threadIdx.x;
  float* row = dout + ((size_t)b * NT + t) * NV;
  for (int l = tid; l < NL; l += 256) {
    atomicAdd(row + enc_in[b * NL + l], score_e[b * 1024 + l]);
  }
}

// ---------------------------------------------------------------- argmax over vocab
__global__ void argmax_kernel(const float* __restrict__ dout, float* __restrict__ syms, int t) {
  int b = blockIdx.x, tid = threadIdx.x;
  const float* row = dout + ((size_t)b * NT + t) * NV;
  float bv = -__builtin_inff();
  int   bi = 0;
  for (int v = tid; v < NV; v += 256) {
    float x = row[v];
    if (x > bv || (x == bv && v < bi)) { bv = x; bi = v; }
  }
  __shared__ float s_v[256];
  __shared__ int   s_i[256];
  s_v[tid] = bv; s_i[tid] = bi; __syncthreads();
  for (int st = 128; st > 0; st >>= 1) {
    if (tid < st) {
      if (s_v[tid + st] > s_v[tid] || (s_v[tid + st] == s_v[tid] && s_i[tid + st] < s_i[tid])) {
        s_v[tid] = s_v[tid + st]; s_i[tid] = s_i[tid + st];
      }
    }
    __syncthreads();
  }
  if (tid == 0) syms[b * NT + t] = (float)s_i[0];
}

// ---------------------------------------------------------------- host
static void* wsp(void* base, size_t& off, size_t bytes) {
  off = (off + 255) & ~(size_t)255;
  void* p = (char*)base + off;
  off += bytes;
  return p;
}

extern "C" void kernel_launch(void* const* d_in, const int* in_sizes, int n_in,
                              void* d_out, int out_size, void* d_ws, size_t ws_size,
                              hipStream_t stream) {
  (void)in_sizes; (void)n_in; (void)out_size; (void)ws_size;
  const int*   enc_in  = (const int*)  d_in[0];
  const float* enc_out = (const float*)d_in[1];
  const unsigned char* enc_mask = (const unsigned char*)d_in[2];
  const int*   inputs  = (const int*)  d_in[3];
  const float* embedW  = (const float*)d_in[4];
  const float* e1_Wih = (const float*)d_in[5];
  const float* e1_Whh = (const float*)d_in[6];
  const float* e1_b   = (const float*)d_in[7];
  const float* e2_Wih = (const float*)d_in[8];
  const float* e2_Whh = (const float*)d_in[9];
  const float* e2_b   = (const float*)d_in[10];
  const float* in_W   = (const float*)d_in[11];
  const float* in_b   = (const float*)d_in[12];
  const float* r1_Wih = (const float*)d_in[13];
  const float* r1_Whh = (const float*)d_in[14];
  const float* r1_b   = (const float*)d_in[15];
  const float* a1_Wq  = (const float*)d_in[16];
  const float* a1_bq  = (const float*)d_in[17];
  const float* a1_Wk  = (const float*)d_in[18];
  const float* a1_bk  = (const float*)d_in[19];
  const float* a1_Wv  = (const float*)d_in[20];
  const float* a1_bv  = (const float*)d_in[21];
  const float* r2_Wih = (const float*)d_in[22];
  const float* r2_Whh = (const float*)d_in[23];
  const float* r2_b   = (const float*)d_in[24];
  const float* a2_Wq  = (const float*)d_in[25];
  const float* a2_bq  = (const float*)d_in[26];
  const float* a2_Wk  = (const float*)d_in[27];
  const float* a2_bk  = (const float*)d_in[28];
  const float* a2_Wv  = (const float*)d_in[29];
  const float* a2_bv  = (const float*)d_in[30];
  const float* copyW  = (const float*)d_in[31];
  const float* copyb  = (const float*)d_in[32];
  const float* outW   = (const float*)d_in[33];
  const float* out_b  = (const float*)d_in[34];
  const float* scale  = (const float*)d_in[35];

  float* dout = (float*)d_out;
  float* syms = dout + (size_t)NB * NT * NV;

  // ---- workspace layout
  size_t off = 0;
  __bf16* encb  = (__bf16*)wsp(d_ws, off, (size_t)NBL * NH * 2);        // 41 MB
  __bf16* w5b   = (__bf16*)wsp(d_ws, off, (size_t)5 * APAD * NH * 2);   // 1.6 MB
  __bf16* outWb = (__bf16*)wsp(d_ws, off, (size_t)NV * KVP * 2);        // 50.5 MB
  __bf16* projb = (__bf16*)wsp(d_ws, off, (size_t)5 * NBL * APAD * 2);  // 128 MB
  size_t state_begin = (off + 255) & ~(size_t)255;
  float* eh1   = (float*)wsp(d_ws, off, (size_t)NB * NH * 4);
  float* ec1   = (float*)wsp(d_ws, off, (size_t)NB * NH * 4);
  float* eh2   = (float*)wsp(d_ws, off, (size_t)NB * NH * 4);
  float* ec2   = (float*)wsp(d_ws, off, (size_t)NB * NH * 4);
  float* h1    = (float*)wsp(d_ws, off, (size_t)NB * NH * 4);
  float* c1    = (float*)wsp(d_ws, off, (size_t)NB * NH * 4);
  float* h2    = (float*)wsp(d_ws, off, (size_t)NB * NH * 4);
  float* c2    = (float*)wsp(d_ws, off, (size_t)NB * NH * 4);
  float* attc  = (float*)wsp(d_ws, off, (size_t)NB * APAD * 4);  // final att carry
  float* att1o = (float*)wsp(d_ws, off, (size_t)NB * APAD * 4);  // attention-1 context
  float* selb  = (float*)wsp(d_ws, off, (size_t)NB * NH * 4);
  float* embB  = (float*)wsp(d_ws, off, (size_t)NB * APAD * 4);
  float* decin = (float*)wsp(d_ws, off, (size_t)NB * NH * 4);
  __bf16* attA = (__bf16*)wsp(d_ws, off, (size_t)48 * KVP * 2);
  float* scE   = (float*)wsp(d_ws, off, (size_t)NB * 1024 * 4);
  size_t state_end = off;

  // zero recurrent state (+ attA pad rows 40..47) every call -> deterministic
  hipMemsetAsync((char*)d_ws + state_begin, 0, state_end - state_begin, stream);

  // ---- one-time conversions + projection GEMM
  cvt_bf16_kernel<<<4096, 256, 0, stream>>>(enc_out, encb, (size_t)NBL * NH);
  pack_w5_kernel <<<1024, 256, 0, stream>>>(a1_Wk, a1_Wv, a2_Wk, a2_Wv, copyW, w5b);
  cvt_outw_kernel<<<4096, 256, 0, stream>>>(outW, outWb);
  {
    int waves = 5 * PMT * NGRP;
    proj_gemm_kernel<<<(waves + 7) / 8, 256, 0, stream>>>(encb, w5b, a1_bk, a1_bv, a2_bk, a2_bv, copyb, projb);
  }

  const size_t PJ = (size_t)NBL * APAD;
  const __bf16* K1 = projb;
  const __bf16* V1 = projb + PJ;
  const __bf16* K2 = projb + 2 * PJ;
  const __bf16* V2 = projb + 3 * PJ;
  const __bf16* CK = projb + 4 * PJ;

  // ---- sequential decode
  for (int t = 0; t < NT; ++t) {
    embed_kernel<<<NB, 256, 0, stream>>>(inputs, embedW, embB, t);
    lstm_kernel <<<NB, 1024, 0, stream>>>(embB, APAD, NE, e1_Wih, e1_Whh, e1_b, eh1, ec1);
    lstm_kernel <<<NB, 1024, 0, stream>>>(eh1,  NH,   NH, e2_Wih, e2_Whh, e2_b, eh2, ec2);
    decin_kernel<<<NB, 512,  0, stream>>>(embB, eh2, attc, selb, in_W, in_b, decin, t);
    lstm_kernel <<<NB, 1024, 0, stream>>>(decin, NH,  NH, r1_Wih, r1_Whh, r1_b, h1, c1);
    attn_kernel <<<NB, 1024, 0, stream>>>(h1, a1_Wq, a1_bq, K1, V1, enc_mask, nullptr, att1o);
    lstm_kernel <<<NB, 1024, 0, stream>>>(att1o, APAD, NA, r2_Wih, r2_Whh, r2_b, h2, c2);
    attn_kernel <<<NB, 1024, 0, stream>>>(h2, a2_Wq, a2_bq, K2, V2, enc_mask, att1o, attc);
    copy_sel_kernel<<<NB, 1024, 0, stream>>>(attc, CK, enc_mask, enc_in, inputs, encb,
                                             scale, scE, selb, attA, t);
    vocab_gemm_kernel<<<(VNT + 7) / 8, 256, 0, stream>>>(attA, outWb, out_b, dout, t);
    scatter_kernel<<<NB, 256, 0, stream>>>(enc_in, scE, dout, t);
    argmax_kernel <<<NB, 256, 0, stream>>>(dout, syms, t);
  }
}
